// _DConvBase_53300544143703
// MI455X (gfx1250) — matrix operand order repaired and checked by reading, not compiled, not run on hardware
//
#include <hip/hip_runtime.h>

typedef __attribute__((ext_vector_type(16))) _Float16 v16h;
typedef __attribute__((ext_vector_type(8)))  _Float16 v8h;
typedef __attribute__((ext_vector_type(8)))  float    v8f;
typedef __attribute__((ext_vector_type(4)))  float    v4f;
typedef __attribute__((ext_vector_type(2)))  float    v2f;

#define CIN   96
#define COUT  96
#define DDIM  8
#define HDIM  48
#define WDIM  48
#define MTOT  (DDIM*HDIM*WDIM)   // 18432 output positions
#define KPTS  27
#define MT    64                 // spatial tile per block
#define OFFC  81                 // 3*27 offset channels

// ---------------------------------------------------------------------------
// Input transpose: [c][spatial] -> [spatial][c]  (channels contiguous)
// ---------------------------------------------------------------------------
__global__ void __launch_bounds__(256)
transpose_in(const float* __restrict__ src, float* __restrict__ dst)
{
    int i = blockIdx.x * 256 + threadIdx.x;   // over 96*18432, m fastest
    if (i >= CIN * MTOT) return;
    int m = i % MTOT;
    int c = i / MTOT;
    dst[m * CIN + c] = src[i];                // coalesced reads over m
}

// Repack main weights fp32 [O][C][27] -> f16 [k][o][c]
__global__ void __launch_bounds__(256)
repack_w_f16(const float* __restrict__ src, _Float16* __restrict__ dst)
{
    int i = blockIdx.x * 256 + threadIdx.x;   // over 27*96*96
    if (i >= KPTS * 96 * 96) return;
    int c = i % 96;
    int o = (i / 96) % 96;
    int k = i / (96 * 96);
    dst[i] = (_Float16)src[(o * CIN + c) * KPTS + k];
}

// Repack offset-conv weights fp32 [81][C][27] -> fp32 [k][o][c], o padded to 96
__global__ void __launch_bounds__(256)
repack_w_f32(const float* __restrict__ src, float* __restrict__ dst)
{
    int i = blockIdx.x * 256 + threadIdx.x;
    if (i >= KPTS * 96 * 96) return;
    int c = i % 96;
    int o = (i / 96) % 96;
    int k = i / (96 * 96);
    dst[i] = (o < OFFC) ? src[(o * CIN + c) * KPTS + k] : 0.0f;
}

// ---------------------------------------------------------------------------
// Pass 1: offset-predicting 3x3x3 conv, full fp32 via V_WMMA_F32_16X16X4_F32.
// ---------------------------------------------------------------------------
__global__ void __launch_bounds__(256)
offconv_kernel(const float* __restrict__ inT,    // [18432][96] transposed input
               const float* __restrict__ Wk,     // [27][96][96] f32 (padded)
               const float* __restrict__ bias,   // [81]
               float*       __restrict__ out)    // [81][18432]
{
    __shared__ float sCols[MT * 96];   // 24 KB col tile, [m][c], fp32
    __shared__ int   sIdx[MT];
    __shared__ float sWt [MT];

    const int tid   = threadIdx.x;
    const int m0    = blockIdx.x * MT;
    const int lane  = tid & 31;
    const int wv    = tid >> 5;
    const int g     = lane >> 4;    // lane half
    const int ln    = lane & 15;    // row/col within 16x16 tile
    const int mt    = wv & 3;       // 16-col slab
    const int ohalf = wv >> 2;      // row half

    v8f acc[3];
    #pragma unroll
    for (int i = 0; i < 3; ++i)
        #pragma unroll
        for (int j = 0; j < 8; ++j) acc[i][j] = 0.0f;

    for (int k = 0; k < KPTS; ++k) {
        const int kz = k / 9, ky = (k / 3) % 3, kx = k % 3;

        if (tid < MT) {
            const int m  = m0 + tid;
            const int xw = m % WDIM;
            const int yh = (m / WDIM) % HDIM;
            const int zd = m / (HDIM * WDIM);
            const int zc = zd - 1 + kz, yc = yh - 1 + ky, xc = xw - 1 + kx;
            const bool ok = (unsigned)zc < (unsigned)DDIM &&
                            (unsigned)yc < (unsigned)HDIM &&
                            (unsigned)xc < (unsigned)WDIM;
            sIdx[tid] = ok ? (zc * HDIM + yc) * WDIM + xc : 0;
            sWt [tid] = ok ? 1.0f : 0.0f;
        }
        __syncthreads();

        {   // build fp32 col tile: 4 lanes cooperate on one position
            const int mm = tid >> 2;          // 0..63
            const int cb = tid & 3;           // 24-channel slab
            const int   i0 = sIdx[mm];
            const float w0 = sWt [mm];
            const v4f* p = (const v4f*)(inT + (size_t)i0 * CIN + cb * 24);
            #pragma unroll
            for (int q = 0; q < 6; ++q)
                *(v4f*)&sCols[mm * 96 + cb * 24 + q * 4] = p[q] * w0;
        }
        __syncthreads();

        // f32 WMMA: 24 chunks of K=4
        const int col = mt * 16 + ln;
        #pragma unroll 4
        for (int cc = 0; cc < 24; ++cc) {
            const int c0 = cc * 4;
            // B 4x16 f32: lane half g holds K = 2g, 2g+1
            const v2f bf = *(const v2f*)&sCols[col * 96 + c0 + g * 2];
            #pragma unroll
            for (int i = 0; i < 3; ++i) {
                const int row = ohalf * 48 + i * 16 + ln;
                // A 16x4 f32: lane half g holds K = 2g, 2g+1
                const v2f af = *(const v2f*)&Wk[(k * 96 + row) * 96 + c0 + g * 2];
                acc[i] = __builtin_amdgcn_wmma_f32_16x16x4_f32(
                    false, af, false, bf, (short)0, acc[i], false, false);
            }
        }
        __syncthreads();
    }

    const int mcol = m0 + mt * 16 + ln;
    #pragma unroll
    for (int i = 0; i < 3; ++i) {
        #pragma unroll
        for (int v = 0; v < 8; ++v) {
            const int o = ohalf * 48 + i * 16 + g * 8 + v;
            if (o < OFFC)
                out[o * MTOT + mcol] = acc[i][v] + bias[o];
        }
    }
}

// ---------------------------------------------------------------------------
// Pass 2: deformable conv as implicit GEMM over (c,k), WMMA f32 <- f16.
// ---------------------------------------------------------------------------
__global__ void __launch_bounds__(256)
deform_kernel(const float*    __restrict__ inT,    // [18432][96] transposed input
              const float*    __restrict__ OFF,    // [81][18432]
              const _Float16* __restrict__ Wk,     // [27][96][96] f16
              const float*    __restrict__ bias,   // [96]
              float*          __restrict__ out)    // [96][18432]
{
    __shared__ _Float16 sCols[MT * 96];   // 12 KB col tile, [m][c]
    __shared__ int      sIdx[MT * 8];     // corner flat spatial indices
    __shared__ float    sWt [MT * 8];     // corner weights (0 if OOB)

    const int tid   = threadIdx.x;
    const int m0    = blockIdx.x * MT;
    const int lane  = tid & 31;
    const int wv    = tid >> 5;
    const int g     = lane >> 4;
    const int ln    = lane & 15;
    const int mt    = wv & 3;
    const int ohalf = wv >> 2;

    v8f acc[3];
    #pragma unroll
    for (int i = 0; i < 3; ++i)
        #pragma unroll
        for (int j = 0; j < 8; ++j) acc[i][j] = 0.0f;

    for (int k = 0; k < KPTS; ++k) {
        const int kz = k / 9, ky = (k / 3) % 3, kx = k % 3;

        // ---- phase 1: trilinear corners per position (64 threads) ----
        if (tid < MT) {
            const int m  = m0 + tid;
            const int xw = m % WDIM;
            const int yh = (m / WDIM) % HDIM;
            const int zd = m / (HDIM * WDIM);
            const float oz = OFF[(k * 3 + 0) * MTOT + m];
            const float oy = OFF[(k * 3 + 1) * MTOT + m];
            const float ox = OFF[(k * 3 + 2) * MTOT + m];
            const float z = (float)(zd - 1 + kz) + oz;
            const float y = (float)(yh - 1 + ky) + oy;
            const float x = (float)(xw - 1 + kx) + ox;
            const float zf = floorf(z), yf = floorf(y), xf = floorf(x);
            const int   z0 = (int)zf, y0 = (int)yf, x0 = (int)xf;
            const float tz = z - zf, ty = y - yf, tx = x - xf;
            #pragma unroll
            for (int t = 0; t < 8; ++t) {
                const int dz = t >> 2, dy = (t >> 1) & 1, dx = t & 1;
                const int zc = z0 + dz, yc = y0 + dy, xc = x0 + dx;
                const bool ok = (unsigned)zc < (unsigned)DDIM &&
                                (unsigned)yc < (unsigned)HDIM &&
                                (unsigned)xc < (unsigned)WDIM;
                const float w = (dz ? tz : 1.0f - tz) *
                                (dy ? ty : 1.0f - ty) *
                                (dx ? tx : 1.0f - tx);
                const int zi = min(max(zc, 0), DDIM - 1);
                const int yi = min(max(yc, 0), HDIM - 1);
                const int xi = min(max(xc, 0), WDIM - 1);
                sIdx[tid * 8 + t] = (zi * HDIM + yi) * WDIM + xi;
                sWt [tid * 8 + t] = ok ? w : 0.0f;
            }
        }
        __syncthreads();

        // ---- phase 2: blend col tile; 4 lanes cooperate on one position ----
        {
            const int mm = tid >> 2;          // 0..63
            const int cb = tid & 3;           // 24-channel slab
            int   ii[8];
            float wt[8];
            #pragma unroll
            for (int t = 0; t < 8; ++t) {
                ii[t] = sIdx[mm * 8 + t];
                wt[t] = sWt [mm * 8 + t];
            }
            float a[24];
            #pragma unroll
            for (int j = 0; j < 24; ++j) a[j] = 0.0f;
            #pragma unroll
            for (int t = 0; t < 8; ++t) {
                const float w = wt[t];
                const v4f* p = (const v4f*)(inT + (size_t)ii[t] * CIN + cb * 24);
                #pragma unroll
                for (int q = 0; q < 6; ++q) {
                    const v4f v = p[q];
                    a[q * 4 + 0] += w * v[0];
                    a[q * 4 + 1] += w * v[1];
                    a[q * 4 + 2] += w * v[2];
                    a[q * 4 + 3] += w * v[3];
                }
            }
            #pragma unroll
            for (int q = 0; q < 3; ++q) {
                v8h h;
                #pragma unroll
                for (int j = 0; j < 8; ++j) h[j] = (_Float16)a[q * 8 + j];
                *(v8h*)&sCols[mm * 96 + cb * 24 + q * 8] = h;
            }
        }
        __syncthreads();

        // ---- phase 3: WMMA, 3 c-chunks of 32 x 3 row tiles per wave ----
        const int col = mt * 16 + ln;
        #pragma unroll
        for (int cc = 0; cc < 3; ++cc) {
            const int c0 = cc * 32;
            // B 32x16 f16: lane half g holds K = g*16 .. g*16+15 (contiguous)
            const v16h bf = *(const v16h*)&sCols[col * 96 + c0 + g * 16];
            #pragma unroll
            for (int i = 0; i < 3; ++i) {
                const int row = ohalf * 48 + i * 16 + ln;
                // A 16x32 f16: lane half g holds K = g*16 .. g*16+15 (contiguous)
                const v8h* pa = (const v8h*)(Wk + ((k * 96 + row) * 96 + c0 + g * 16));
                const v8h alo = pa[0];
                const v8h ahi = pa[1];      // +8 halves
                const v16h af = __builtin_shufflevector(
                    alo, ahi, 0, 1, 2, 3, 4, 5, 6, 7, 8, 9, 10, 11, 12, 13, 14, 15);
                acc[i] = __builtin_amdgcn_wmma_f32_16x16x32_f16(
                    false, af, false, bf, (short)0, acc[i], false, false);
            }
        }
        __syncthreads();
    }

    const int mcol = m0 + mt * 16 + ln;
    #pragma unroll
    for (int i = 0; i < 3; ++i) {
        #pragma unroll
        for (int v = 0; v < 8; ++v) {
            const int o = ohalf * 48 + i * 16 + g * 8 + v;
            out[o * MTOT + mcol] = acc[i][v] + bias[o];
        }
    }
}

// ---------------------------------------------------------------------------
extern "C" void kernel_launch(void* const* d_in, const int* in_sizes, int n_in,
                              void* d_out, int out_size, void* d_ws, size_t ws_size,
                              hipStream_t stream)
{
    const float* inp    = (const float*)d_in[0];   // [96][8][48][48]
    const float* w_off  = (const float*)d_in[1];   // [81][96][27]
    const float* b_off  = (const float*)d_in[2];   // [81]
    const float* weight = (const float*)d_in[3];   // [96][96][27]
    const float* bias   = (const float*)d_in[4];   // [96]
    float* out = (float*)d_out;                    // [96][18432]

    char* ws = (char*)d_ws;
    size_t off0 = 0;                                   // OFF   : 81*18432 f32
    size_t off1 = off0 + (size_t)OFFC * MTOT * 4;      // wMain : 27*96*96 f16
    size_t off2 = off1 + (size_t)KPTS * 96 * 96 * 2;   // wOffF : 27*96*96 f32
    size_t off3 = off2 + (size_t)KPTS * 96 * 96 * 4;   // inT   : 18432*96 f32
    float*    OFF   = (float*)(ws + off0);
    _Float16* wMain = (_Float16*)(ws + off1);
    float*    wOffF = (float*)(ws + off2);
    float*    inT   = (float*)(ws + off3);

    const int ntr = CIN * MTOT;
    transpose_in<<<(ntr + 255) / 256, 256, 0, stream>>>(inp, inT);
    const int nrep = KPTS * 96 * 96;
    repack_w_f16<<<(nrep + 255) / 256, 256, 0, stream>>>(weight, wMain);
    repack_w_f32<<<(nrep + 255) / 256, 256, 0, stream>>>(w_off,  wOffF);

    const int nblk = MTOT / MT;   // 288
    // pass 1: offset-predicting conv (full fp32 WMMA) -> OFF (+ b_off)
    offconv_kernel<<<nblk, 256, 0, stream>>>(inT, wOffF, b_off, OFF);
    // pass 2: deformable conv consuming OFF -> final output (+ bias)
    deform_kernel<<<nblk, 256, 0, stream>>>(inT, OFF, wMain, bias, out);
}
